// BasicBlock_XNOR_52355651338898
// MI455X (gfx1250) — 8-bit matrix operand order repaired and checked by reading, compiled, not run on hardware
//
#include <hip/hip_runtime.h>
#include <cstdint>
#include <cstddef>

typedef __attribute__((ext_vector_type(8))) int v8i;

#define CCH   128
#define HH    56
#define WW    56
#define NIMG  64
#define HWP   (HH*WW)
#define NPIX  (NIMG*HWP)
#define TOT   (NPIX*CCH)
#define KSTEPS 18
#define NT     8
#define KW_BYTES (KSTEPS*NT*1024)
#define EPSV  1e-5f

__global__ __launch_bounds__(256)
void stats_kernel(const float* __restrict__ x, float* __restrict__ sums) {
    __shared__ float s1[256];
    __shared__ float s2[256];
    const int c = blockIdx.x;
    float sum = 0.f, sq = 0.f;
    for (int i = threadIdx.x; i < NIMG * HWP; i += 256) {
        const int n = i / HWP;
        const int r = i - n * HWP;
        const float v = x[(size_t)(n * CCH + c) * HWP + r];
        sum += v; sq += v * v;
    }
    s1[threadIdx.x] = sum; s2[threadIdx.x] = sq;
    __syncthreads();
    for (int o = 128; o > 0; o >>= 1) {
        if (threadIdx.x < o) {
            s1[threadIdx.x] += s1[threadIdx.x + o];
            s2[threadIdx.x] += s2[threadIdx.x + o];
        }
        __syncthreads();
    }
    if (threadIdx.x == 0) { sums[c] = s1[0]; sums[CCH + c] = s2[0]; }
}

__global__ __launch_bounds__(256)
void prepw_kernel(const float* __restrict__ w, float* __restrict__ scale,
                  int8_t* __restrict__ wsw) {
    __shared__ float red[256];
    const int oc = blockIdx.x;
    float s = 0.f;
    for (int j = threadIdx.x; j < 1152; j += 256)
        s += fabsf(w[oc * 1152 + j]);
    red[threadIdx.x] = s;
    __syncthreads();
    for (int o = 128; o > 0; o >>= 1) {
        if (threadIdx.x < o) red[threadIdx.x] += red[threadIdx.x + o];
        __syncthreads();
    }
    if (threadIdx.x == 0) scale[oc] = red[0] * (1.f / 1152.f);

    const int ntile = oc >> 4;
    for (int j = threadIdx.x; j < 1152; j += 256) {
        const int sstep = j >> 6;
        const int kk    = j & 63;
        const int hi    = kk >> 5;
        const int kl    = kk & 31;
        const int half  = (kl >> 4) & 1;
        const int v     = (hi ? 4 : 0) + ((kl & 15) >> 2);
        const int b     = kl & 3;
        const int lane  = half * 16 + (oc & 15);
        const int ic    = (sstep & 1) * 64 + kk;
        const int tap   = sstep >> 1;
        const float wv  = w[oc * 1152 + ic * 9 + tap];
        const int8_t sg = (int8_t)((wv > 0.f) - (wv < 0.f));
        wsw[((sstep * NT + ntile) * 1024) + lane * 32 + v * 4 + b] = sg;
    }
}

__global__ __launch_bounds__(256)
void binact_kernel(const float* __restrict__ x, const float* __restrict__ sums,
                   const float* __restrict__ g, const float* __restrict__ bta,
                   int8_t* __restrict__ act) {
    const size_t i = (size_t)blockIdx.x * 256 + threadIdx.x;
    if (i >= (size_t)TOT) return;
    const int n   = (int)(i / (CCH * HWP));
    const int c   = (int)((i / HWP) % CCH);
    const int pix = (int)(i % HWP);
    const float cnt  = (float)(NIMG * HWP);
    const float mean = sums[c] / cnt;
    const float var  = sums[CCH + c] / cnt - mean * mean;
    const float inv  = rsqrtf(var + EPSV);
    const float v = (x[i] - mean) * inv * g[c] + bta[c];
    act[((size_t)n * HWP + pix) * CCH + c] = (int8_t)((v > 0.f) - (v < 0.f));
}

__global__ __launch_bounds__(256)
void binconv_kernel(const int8_t* __restrict__ act,
                    const int8_t* __restrict__ wsw,
                    const float* __restrict__ scale,
                    const float* __restrict__ prelu_a,
                    float* __restrict__ y) {
    __shared__ int8_t ldsW[KW_BYTES];

    {
        const uint32_t lbase = (uint32_t)(size_t)(void*)ldsW;
        for (int i = threadIdx.x; i < KW_BYTES / 16; i += 256) {
            const uint32_t loff = lbase + (uint32_t)i * 16u;
            const int8_t* gsrc = wsw + (size_t)i * 16;
            asm volatile("global_load_async_to_lds_b128 %0, %1, off"
                         :: "v"(loff), "v"(gsrc) : "memory");
        }
        asm volatile("s_wait_asynccnt 0x0" ::: "memory");
    }
    __syncthreads();

    const int tid  = threadIdx.x;
    const int wave = tid >> 5;
    const int lane = tid & 31;
    const int tile = blockIdx.x * 8 + wave;
    const int mrow = lane & 15;
    const int half = lane >> 4;

    const int p    = tile * 16 + mrow;
    const int nimg = p / HWP;
    const int rem  = p - nimg * HWP;
    const int oh   = rem / WW;
    const int ow   = rem - oh * WW;

    v8i acc[NT];
    const v8i vzero = {0, 0, 0, 0, 0, 0, 0, 0};
#pragma unroll
    for (int t = 0; t < NT; ++t) acc[t] = vzero;

    const float ap = *prelu_a;

    for (int s = 0; s < KSTEPS; ++s) {
        const int tap   = s >> 1;
        const int kh    = tap / 3;
        const int kw    = tap - kh * 3;
        const int icoff = (s & 1) * 64;
        const int ih    = oh + kh - 1;
        const int iw    = ow + kw - 1;
        const bool valid = (ih >= 0) & (ih < HH) & (iw >= 0) & (iw < WW);
        const size_t poff = ((size_t)(nimg * HWP + ih * WW + iw)) * CCH + icoff;
        const int8_t* base = act + (valid ? poff : 0);
        const int hb = half * 16;
        int2 a01 = *(const int2*)(base + hb);
        int2 a23 = *(const int2*)(base + 8 + hb);
        int2 a45 = *(const int2*)(base + 32 + hb);
        int2 a67 = *(const int2*)(base + 40 + hb);
        const int msk = valid ? ~0 : 0;
        v8i afrag;
        afrag[0] = a01.x & msk; afrag[1] = a01.y & msk;
        afrag[2] = a23.x & msk; afrag[3] = a23.y & msk;
        afrag[4] = a45.x & msk; afrag[5] = a45.y & msk;
        afrag[6] = a67.x & msk; afrag[7] = a67.y & msk;

#pragma unroll
        for (int t = 0; t < NT; ++t) {
            const v8i bfrag =
                *(const v8i*)(ldsW + (size_t)(s * NT + t) * 1024 + lane * 32);
            acc[t] = __builtin_amdgcn_wmma_i32_16x16x64_iu8(
                true, afrag, true, bfrag, acc[t], false, false);
        }
    }

    int ni8[8], rr8[8];
#pragma unroll
    for (int r = 0; r < 8; ++r) {
        const int m  = r + half * 8;
        const int pp = tile * 16 + m;
        ni8[r] = pp / HWP;
        rr8[r] = pp - ni8[r] * HWP;
    }
#pragma unroll
    for (int t = 0; t < NT; ++t) {
        const int oc = t * 16 + (lane & 15);
        const float sc = scale[oc];
#pragma unroll
        for (int r = 0; r < 8; ++r) {
            float f = sc * (float)acc[t][r];
            f = (f >= 0.f) ? f : ap * f;
            y[(size_t)(ni8[r] * CCH + oc) * HWP + rr8[r]] = f;
        }
    }
}

__global__ __launch_bounds__(256)
void final_kernel(const float* __restrict__ y2, const float* __restrict__ x,
                  const float* __restrict__ sums, const float* __restrict__ g,
                  const float* __restrict__ bta, const float* __restrict__ a3,
                  float* __restrict__ out) {
    const size_t i = (size_t)blockIdx.x * 256 + threadIdx.x;
    if (i >= (size_t)TOT) return;
    const int c = (int)((i / HWP) % CCH);
    const float cnt  = (float)(NIMG * HWP);
    const float mean = sums[c] / cnt;
    const float var  = sums[CCH + c] / cnt - mean * mean;
    const float inv  = rsqrtf(var + EPSV);
    float v = (y2[i] - mean) * inv * g[c] + bta[c] + x[i];
    const float a = *a3;
    out[i] = (v >= 0.f) ? v : a * v;
}

extern "C" void kernel_launch(void* const* d_in, const int* in_sizes, int n_in,
                              void* d_out, int out_size, void* d_ws, size_t ws_size,
                              hipStream_t stream) {
    const float* x    = (const float*)d_in[0];
    const float* g1   = (const float*)d_in[1];
    const float* b1   = (const float*)d_in[2];
    const float* w1   = (const float*)d_in[3];
    const float* a1   = (const float*)d_in[4];
    const float* g2   = (const float*)d_in[5];
    const float* b2   = (const float*)d_in[6];
    const float* w2   = (const float*)d_in[7];
    const float* a2   = (const float*)d_in[8];
    const float* g3   = (const float*)d_in[9];
    const float* b3   = (const float*)d_in[10];
    const float* a3   = (const float*)d_in[11];
    float* out = (float*)d_out;

    uint8_t* ws = (uint8_t*)d_ws;
    int8_t* act   = (int8_t*)ws;
    float*  y     = (float*)(ws + (size_t)TOT);
    int8_t* wsw1  = (int8_t*)(ws + (size_t)TOT * 5);
    int8_t* wsw2  = wsw1 + KW_BYTES;
    float*  sc1   = (float*)(wsw2 + KW_BYTES);
    float*  sc2   = sc1 + CCH;
    float*  sums1 = sc2 + CCH;
    float*  sums2 = sums1 + 2 * CCH;
    float*  sums3 = sums2 + 2 * CCH;

    const int ew_grid = (TOT + 255) / 256;
    const int cv_grid = (NPIX / 16) / 8;

    prepw_kernel  <<<CCH, 256, 0, stream>>>(w1, sc1, wsw1);
    prepw_kernel  <<<CCH, 256, 0, stream>>>(w2, sc2, wsw2);
    stats_kernel  <<<CCH, 256, 0, stream>>>(x, sums1);
    binact_kernel <<<ew_grid, 256, 0, stream>>>(x, sums1, g1, b1, act);
    binconv_kernel<<<cv_grid, 256, 0, stream>>>(act, wsw1, sc1, a1, y);
    stats_kernel  <<<CCH, 256, 0, stream>>>(y, sums2);
    binact_kernel <<<ew_grid, 256, 0, stream>>>(y, sums2, g2, b2, act);
    binconv_kernel<<<cv_grid, 256, 0, stream>>>(act, wsw2, sc2, a2, y);
    stats_kernel  <<<CCH, 256, 0, stream>>>(y, sums3);
    final_kernel  <<<ew_grid, 256, 0, stream>>>(y, x, sums3, g3, b3, a3, out);
}
